// MultiHeadAttention_44856638439987
// MI455X (gfx1250) — matrix operand order repaired and checked by reading, not compiled, not run on hardware
//
#include <hip/hip_runtime.h>

// ---------------------------------------------------------------------------
// MI455X (gfx1250) multi-head attention, bf16 WMMA pipeline.
//   B=4, S=2048, D=1024, H=16, hd=64   (~137 GFLOP, compute-bound)
// Pipeline:
//   0) one-time fp32->bf16 convert of inputs + weights (HBM ~8us, removes all
//      per-iteration cvt VALU from the GEMM hot loops)
//   1) q = queries @ Wq^T   (bf16 out, natural layout)      [wmma bf16]
//   2) k = keys    @ Wk^T   (bf16 out, natural layout)      [wmma bf16]
//   3) vT = (values @ Wv^T) transposed per head (d-major)   [wmma bf16]
//   4) attention core: scores in LDS, softmax, P@V          [wmma bf16]
//   5) out = attnOut @ Wo^T (f32 out)                       [wmma bf16]
// All bf16 intermediates (~120 MB) are L2-resident (192 MB L2), so WMMA
// fragments load straight from global; LDS is used where data crosses
// lanes/waves (score strip, softmax, epilogue transpose, reductions).
// ---------------------------------------------------------------------------

typedef __attribute__((ext_vector_type(16))) __bf16 v16bf;
typedef __attribute__((ext_vector_type(8)))  __bf16 v8bf;
typedef __attribute__((ext_vector_type(4)))  __bf16 v4bf;
typedef __attribute__((ext_vector_type(8)))  float  v8f;
typedef __attribute__((ext_vector_type(4)))  float  v4f;

static constexpr int BATCH = 4;
static constexpr int SEQ   = 2048;
static constexpr int DIM   = 1024;
static constexpr int HEADS = 16;
static constexpr int HDIM  = 64;
static constexpr int MROWS = BATCH * SEQ;      // 8192
static constexpr float NEGMASK = -1000000.0f;

// ----- fp32 -> bf16 streaming convert ----------------------------------------

__global__ __launch_bounds__(256) void convert_bf16(const float* __restrict__ src,
                                                    __bf16* __restrict__ dst,
                                                    int n8 /* n / 8 */) {
  int i = blockIdx.x * blockDim.x + threadIdx.x;
  int stride = gridDim.x * blockDim.x;
  for (; i < n8; i += stride) {
    v4f a = *(const v4f*)(src + (size_t)i * 8);
    v4f b = *(const v4f*)(src + (size_t)i * 8 + 4);
    v8bf o;
#pragma unroll
    for (int j = 0; j < 4; ++j) { o[j] = (__bf16)a[j]; o[4 + j] = (__bf16)b[j]; }
    *(v8bf*)(dst + (size_t)i * 8) = o;
  }
}

// ----- WMMA fragment loaders --------------------------------------------------
// A fragment (16x32 MxK bf16): lane L holds row M=L&15; K elems kbase+{0..7}
// in elems 0..7 and kbase+{16..23} in elems 8..15; kbase=8 for lanes 16..31
// (ISA 7.12.2 16-bit A layout).

__device__ __forceinline__ v16bf a_frag(const __bf16* __restrict__ base, int ld, int k0) {
  int lane = threadIdx.x & 31;
  int m  = lane & 15;
  int kb = k0 + ((lane & 16) ? 8 : 0);
  const __bf16* p = base + (size_t)m * ld + kb;
  v8bf lo = *(const v8bf*)(p);        // K kb+0..7
  v8bf hi = *(const v8bf*)(p + 16);   // K kb+16..23
  v16bf r;
#pragma unroll
  for (int i = 0; i < 8; ++i) { r[i] = lo[i]; r[8 + i] = hi[i]; }
  return r;
}

// A fragment built from an f32 source (used for the LDS probability strip).
__device__ __forceinline__ v16bf a_frag_f32(const float* __restrict__ base, int ld, int k0) {
  int lane = threadIdx.x & 31;
  int m  = lane & 15;
  int kb = k0 + ((lane & 16) ? 8 : 0);
  const float* p = base + (size_t)m * ld + kb;
  v4f x0 = *(const v4f*)(p + 0);
  v4f x1 = *(const v4f*)(p + 4);
  v4f x2 = *(const v4f*)(p + 16);
  v4f x3 = *(const v4f*)(p + 20);
  v16bf r;
#pragma unroll
  for (int i = 0; i < 4; ++i) {
    r[i]      = (__bf16)x0[i];
    r[4 + i]  = (__bf16)x1[i];
    r[8 + i]  = (__bf16)x2[i];
    r[12 + i] = (__bf16)x3[i];
  }
  return r;
}

// B fragment (32x16 KxN bf16): lane L holds column N=L&15; K elems
// kb+{0..7} in elems 0..7 and kb+{16..23} in elems 8..15, kb = k0 + (L<16?0:8). Source is row-major over K at
// fixed N (B[k][n] = src[n*ld + k]) — matches W(N,K) for x@W^T, K rows for
// q@k^T, and d-major vT for P@V, all with zero data reshuffling.

__device__ __forceinline__ v16bf b_frag(const __bf16* __restrict__ base, int ld, int k0) {
  int lane = threadIdx.x & 31;
  int n  = lane & 15;
  int kk = k0 + ((lane & 16) ? 8 : 0);
  const __bf16* p = base + (size_t)n * ld + kk;
  v8bf lo = *(const v8bf*)(p);
  v8bf hi = *(const v8bf*)(p + 16);
  v16bf r;
#pragma unroll
  for (int i = 0; i < 8; ++i) { r[i] = lo[i]; r[8 + i] = hi[i]; }
  return r;
}

#define WMMA_BF16(a, b, c) \
  __builtin_amdgcn_wmma_f32_16x16x32_bf16(false, (a), false, (b), (short)0, (c), false, false)

// ----- Projection GEMM:  Y(MxN) = A(MxK) @ W(NxK)^T ---------------------------
// Block = 256 threads (8 waves). Block tile 256(M) x 64(N); wave tile 32x64:
// 2 A-frags x 4 B-frags -> 8 *independent* WMMAs per K-step (hides the
// WMMA->WMMA/VALU hazard slots and memory latency).
// MODE 0: bf16 natural-layout out (LDS-transposed coalesced stores).
// MODE 1: bf16 per-head-transposed out vT[(b*H+h)*HD + d][s] (a C-fragment
//         lane holds 8 consecutive M at fixed N -> contiguous b128 stores).
// MODE 2: f32 natural-layout out (final projection).

template <int MODE>
__global__ __launch_bounds__(256) void gemm_wmma(const __bf16* __restrict__ A,
                                                 const __bf16* __restrict__ W,
                                                 void* __restrict__ out,
                                                 int M, int N, int K) {
  const int w    = threadIdx.x >> 5;
  const int lane = threadIdx.x & 31;
  const int m0   = blockIdx.y * 256 + w * 32;
  const int n0   = blockIdx.x * 64;
  const int hi   = lane >> 4;

  const __bf16* A0 = A + (size_t)m0 * K;
  const __bf16* A1 = A + (size_t)(m0 + 16) * K;

  v8f acc[2][4] = {};
  for (int kk = 0; kk < K; kk += 32) {
    v16bf a0 = a_frag(A0, K, kk);
    v16bf a1 = a_frag(A1, K, kk);
    __builtin_prefetch(A0 + (size_t)(lane & 15) * K + kk + 64, 0, 1);  // global_prefetch_b8
#pragma unroll
    for (int t = 0; t < 4; ++t) {
      v16bf b = b_frag(W + (size_t)(n0 + 16 * t) * K, K, kk);
      acc[0][t] = WMMA_BF16(a0, b, acc[0][t]);
      acc[1][t] = WMMA_BF16(a1, b, acc[1][t]);
    }
  }

  if constexpr (MODE == 0) {
    __shared__ __bf16 stage[8][32][72];           // pad 72 vs bank conflicts
#pragma unroll
    for (int st = 0; st < 2; ++st)
#pragma unroll
      for (int t = 0; t < 4; ++t) {
        int nn = 16 * t + (lane & 15);
#pragma unroll
        for (int r = 0; r < 8; ++r)
          stage[w][16 * st + r + 8 * hi][nn] = (__bf16)acc[st][t][r];
      }
    __syncthreads();
    __bf16* Y = (__bf16*)out;
    const __bf16* srow = &stage[w][lane][0];       // one row (64 bf16) per lane
    __bf16* drow = Y + (size_t)(m0 + lane) * N + n0;
#pragma unroll
    for (int i = 0; i < 8; ++i) *(v8bf*)(drow + 8 * i) = *(const v8bf*)(srow + 8 * i);
  } else if constexpr (MODE == 1) {
    int bb = m0 / SEQ;
#pragma unroll
    for (int st = 0; st < 2; ++st) {
      int s0 = (m0 % SEQ) + 16 * st + 8 * hi;      // 8 consecutive seq rows
#pragma unroll
      for (int t = 0; t < 4; ++t) {
        int ng = n0 + 16 * t + (lane & 15);
        int hh = ng >> 6, dh = ng & 63;
        v8bf pk;
#pragma unroll
        for (int r = 0; r < 8; ++r) pk[r] = (__bf16)acc[st][t][r];
        __bf16* dst = (__bf16*)out +
                      ((size_t)((bb * HEADS + hh) * HDIM + dh)) * SEQ + s0;
        *(v8bf*)dst = pk;
      }
    }
  } else {  // MODE 2: f32 natural layout
    float* Y = (float*)out;
#pragma unroll
    for (int st = 0; st < 2; ++st)
#pragma unroll
      for (int t = 0; t < 4; ++t) {
        int ng = n0 + 16 * t + (lane & 15);
#pragma unroll
        for (int r = 0; r < 8; ++r)
          Y[(size_t)(m0 + 16 * st + r + 8 * hi) * N + ng] = acc[st][t][r];
      }
  }
}

// ----- Attention core ---------------------------------------------------------
// One block: 16 query rows of one (b,h). 8 waves split the 2048 keys.
// LDS: score strip 16x2048 f32 (128 KB) + reduction buffers — fits 320 KB WGP
// LDS. Softmax in-place in the strip; P@V wave partials merge via ds_add_f32.

__global__ __launch_bounds__(256) void attn_kernel(const __bf16* __restrict__ q,
                                                   const __bf16* __restrict__ k,
                                                   const __bf16* __restrict__ vT,
                                                   const int* __restrict__ valid_len,
                                                   __bf16* __restrict__ attnOut) {
  extern __shared__ char smem[];
  float* scoreBuf = (float*)smem;                  // [16][2048]
  float* red      = scoreBuf + 16 * 2048;          // [16][16]
  float* outBuf   = red + 16 * 16;                 // [16][64]

  const int tid = threadIdx.x, w = tid >> 5, lane = tid & 31, hi = lane >> 4;
  const int qb = blockIdx.x * 16, h = blockIdx.y, b = blockIdx.z;
  const int vlen = valid_len[b];

  const __bf16* qbase = q + ((size_t)(b * SEQ + qb)) * DIM + h * HDIM;
  const __bf16* kbase = k + ((size_t)(b * SEQ)) * DIM + h * HDIM;
  const __bf16* vTh   = vT + (size_t)(b * HEADS + h) * HDIM * SEQ;

  for (int i = tid; i < 16 * 64; i += 256) outBuf[i] = 0.0f;

  // -- Phase 1: scores = (q @ k^T)/sqrt(hd), key-padding mask, -> LDS.
  //    Two key tiles per iteration -> two independent WMMA accumulator chains.
  v16bf aq0 = a_frag(qbase, DIM, 0);
  v16bf aq1 = a_frag(qbase, DIM, 32);
  for (int jt = 0; jt < 16; jt += 2) {
    int j0 = w * 256 + jt * 16;
    v8f acc0 = {}, acc1 = {};
    acc0 = WMMA_BF16(aq0, b_frag(kbase + (size_t)j0 * DIM, DIM, 0), acc0);
    acc1 = WMMA_BF16(aq0, b_frag(kbase + (size_t)(j0 + 16) * DIM, DIM, 0), acc1);
    acc0 = WMMA_BF16(aq1, b_frag(kbase + (size_t)j0 * DIM, DIM, 32), acc0);
    acc1 = WMMA_BF16(aq1, b_frag(kbase + (size_t)(j0 + 16) * DIM, DIM, 32), acc1);
    int n0c = j0 + (lane & 15);
    int n1c = n0c + 16;
    bool m0c = (n0c >= vlen), m1c = (n1c >= vlen);
#pragma unroll
    for (int r = 0; r < 8; ++r) {
      int mm = r + 8 * hi;
      scoreBuf[(size_t)mm * 2048 + n0c] = m0c ? NEGMASK : acc0[r] * 0.125f;
      scoreBuf[(size_t)mm * 2048 + n1c] = m1c ? NEGMASK : acc1[r] * 0.125f;
    }
  }
  __syncthreads();

  // -- Phase 2: row softmax over 2048 keys (16 threads per row)
  const int row = tid >> 4, seg = tid & 15;
  float* srow = scoreBuf + (size_t)row * 2048;
  float lmax = -3.4e38f;
  for (int j = seg * 128; j < seg * 128 + 128; ++j) lmax = fmaxf(lmax, srow[j]);
  red[row * 16 + seg] = lmax;
  __syncthreads();
  if (seg == 0) {
    float m = red[row * 16];
    for (int i = 1; i < 16; ++i) m = fmaxf(m, red[row * 16 + i]);
    red[row * 16] = m;
  }
  __syncthreads();
  const float rmax = red[row * 16];
  __syncthreads();
  float lsum = 0.0f;
  for (int j = seg * 128; j < seg * 128 + 128; ++j) {
    float e = __expf(srow[j] - rmax);
    srow[j] = e;
    lsum += e;
  }
  red[row * 16 + seg] = lsum;
  __syncthreads();
  if (seg == 0) {
    float s = 0.0f;
    for (int i = 0; i < 16; ++i) s += red[row * 16 + i];
    red[row * 16] = s;
  }
  __syncthreads();
  const float inv = 1.0f / red[row * 16];
  for (int j = seg * 128; j < seg * 128 + 128; ++j) srow[j] *= inv;
  __syncthreads();

  // -- Phase 3: out(16x64) = P @ V ; each wave handles 256 of the K=2048 keys;
  //    4 independent accumulator chains per K-step.
  v8f o[4] = {};
  for (int step = 0; step < 8; ++step) {
    int j0 = w * 256 + step * 32;
    v16bf a = a_frag_f32(scoreBuf, 2048, j0);
#pragma unroll
    for (int t = 0; t < 4; ++t) {
      v16bf bf = b_frag(vTh + (size_t)(16 * t) * SEQ, SEQ, j0);
      o[t] = WMMA_BF16(a, bf, o[t]);
    }
  }
#pragma unroll
  for (int t = 0; t < 4; ++t) {
    int n = 16 * t + (lane & 15);
#pragma unroll
    for (int r = 0; r < 8; ++r)
      atomicAdd(&outBuf[(r + 8 * hi) * 64 + n], o[t][r]);   // ds_add_f32
  }
  __syncthreads();

  // -- Phase 4: bf16 store, natural (b, s, h*64+d) layout for the O-projection
  const int m = tid >> 4, n0 = (tid & 15) * 4;
  v4bf pk;
#pragma unroll
  for (int i = 0; i < 4; ++i) pk[i] = (__bf16)outBuf[m * 64 + n0 + i];
  __bf16* dst = attnOut + ((size_t)(b * SEQ + qb + m)) * DIM + h * HDIM + n0;
  *(v4bf*)dst = pk;
}

// ----- Host-side launcher -----------------------------------------------------

extern "C" void kernel_launch(void* const* d_in, const int* in_sizes, int n_in,
                              void* d_out, int out_size, void* d_ws, size_t ws_size,
                              hipStream_t stream) {
  (void)in_sizes; (void)n_in; (void)out_size; (void)ws_size;
  const float* queries = (const float*)d_in[0];
  const float* keys    = (const float*)d_in[1];
  const float* values  = (const float*)d_in[2];
  const int*   vlen    = (const int*)d_in[3];
  const float* Wqf     = (const float*)d_in[4];
  const float* Wkf     = (const float*)d_in[5];
  const float* Wvf     = (const float*)d_in[6];
  const float* Wof     = (const float*)d_in[7];
  float* out = (float*)d_out;

  constexpr size_t XN = (size_t)MROWS * DIM;   // 8M elems
  constexpr size_t WN = (size_t)DIM * DIM;     // 1M elems

  // Workspace (bf16), all L2-resident (~120 MB total):
  __bf16* ws   = (__bf16*)d_ws;
  __bf16* xq   = ws;            // queries bf16
  __bf16* xk   = xq + XN;       // keys bf16
  __bf16* xv   = xk + XN;       // values bf16
  __bf16* wq   = xv + XN;       // weights bf16
  __bf16* wk   = wq + WN;
  __bf16* wv   = wk + WN;
  __bf16* wo   = wv + WN;
  __bf16* qbuf = wo + WN;       // projected q (natural)
  __bf16* kbuf = qbuf + XN;     // projected k (natural)
  __bf16* vT   = kbuf + XN;     // projected v (per-head d-major)
  __bf16* aout = vT + XN;       // attention output (natural)

  dim3 blk(256);

  convert_bf16<<<2048, blk, 0, stream>>>(queries, xq, (int)(XN / 8));
  convert_bf16<<<2048, blk, 0, stream>>>(keys,    xk, (int)(XN / 8));
  convert_bf16<<<2048, blk, 0, stream>>>(values,  xv, (int)(XN / 8));
  convert_bf16<<<512,  blk, 0, stream>>>(Wqf, wq, (int)(WN / 8));
  convert_bf16<<<512,  blk, 0, stream>>>(Wkf, wk, (int)(WN / 8));
  convert_bf16<<<512,  blk, 0, stream>>>(Wvf, wv, (int)(WN / 8));
  convert_bf16<<<512,  blk, 0, stream>>>(Wof, wo, (int)(WN / 8));

  dim3 gg(DIM / 64, MROWS / 256);
  gemm_wmma<0><<<gg, blk, 0, stream>>>(xq, wq, (void*)qbuf, MROWS, DIM, DIM);
  gemm_wmma<0><<<gg, blk, 0, stream>>>(xk, wk, (void*)kbuf, MROWS, DIM, DIM);
  gemm_wmma<1><<<gg, blk, 0, stream>>>(xv, wv, (void*)vT,   MROWS, DIM, DIM);

  dim3 ga(SEQ / 16, HEADS, BATCH);
  size_t smem = (size_t)(16 * 2048 + 16 * 16 + 16 * 64) * sizeof(float);
  attn_kernel<<<ga, blk, smem, stream>>>(qbuf, kbuf, vT, vlen, aout);

  gemm_wmma<2><<<gg, blk, 0, stream>>>(aout, wo, (void*)out, MROWS, DIM, DIM);
}
